// Model_39676907882252
// MI455X (gfx1250) — matrix operand order repaired and checked by reading, not compiled, not run on hardware
//
#include <hip/hip_runtime.h>

typedef __attribute__((ext_vector_type(16))) _Float16 v16h;
typedef __attribute__((ext_vector_type(8)))  _Float16 v8h;
typedef __attribute__((ext_vector_type(8)))  float    v8f;
typedef __attribute__((ext_vector_type(4)))  float    f32x4;
typedef int      v4i  __attribute__((vector_size(16)));   // matches builtin protos
typedef int      v8i  __attribute__((vector_size(32)));
typedef unsigned v4u  __attribute__((vector_size(16)));
typedef __attribute__((address_space(3))) v4i* lds_v4i_t;

#define S_LEN   2048
#define D_DIM   64
#define KT      64     // keys per iteration (4 groups of 16)
#define QT_BLK  128    // q rows per block (8 waves * 16)
#define NWAVE   8
#define N_TILES (S_LEN / KT)

#define WMMA_F16(A, B, C) \
    __builtin_amdgcn_wmma_f32_16x16x32_f16(false, (A), false, (B), (short)0, (C), false, false)

// ---------------- gfx1250 async global->LDS path (guarded) ----------------
#if __has_builtin(__builtin_amdgcn_global_load_async_to_lds_b128)
#define ASYNC_LDS 1
#else
#define ASYNC_LDS 0
#endif

// ---------------- gfx1250 Tensor Data Mover path (guarded) ----------------
#if __has_builtin(__builtin_amdgcn_tensor_load_to_lds)
#define USE_TDM 1
#else
#define USE_TDM 0
#endif

__device__ __forceinline__ void async_copy_b128(const f32x4* g, f32x4* l) {
#if ASYNC_LDS
    __builtin_amdgcn_global_load_async_to_lds_b128((v4i*)g, (lds_v4i_t)l, 0, 0);
#else
    *l = *g;   // synchronous fallback
#endif
}

__device__ __forceinline__ void async_wait0() {
#if ASYNC_LDS
#if __has_builtin(__builtin_amdgcn_s_wait_asynccnt)
    __builtin_amdgcn_s_wait_asynccnt(0);
#else
    asm volatile("s_wait_asynccnt 0" ::: "memory");
#endif
#endif
}

#if USE_TDM
__device__ __forceinline__ unsigned lds_addr_of(const void* p) {
    return (unsigned)(unsigned long long)
        (__attribute__((address_space(3))) const void*)p;
}

// One 2-D TDM descriptor: 64x64 fp32 tile, contiguous rows (stride 64).
__device__ __forceinline__ void tdm_load_tile(const float* gsrc, float* ldst) {
    unsigned long long ga = (unsigned long long)gsrc;
    v4u g0 = {0, 0, 0, 0};
    g0[0] = 1u;                                          // count=1, user D#
    g0[1] = lds_addr_of(ldst);                           // lds_addr (bytes)
    g0[2] = (unsigned)(ga & 0xFFFFFFFFu);                // global_addr[31:0]
    g0[3] = (unsigned)((ga >> 32) & 0x01FFFFFFu)         // global_addr[56:32]
          | (2u << 30);                                  // type = 2 ("image")
    v8i g1 = {0, 0, 0, 0, 0, 0, 0, 0};
    g1[0] = (int)(2u << 16);                             // data_size = 4 bytes
    g1[1] = (int)(64u << 16);                            // tensor_dim0 = 64  (bits 79:48 lo)
    g1[2] = (int)(64u << 16);                            // tensor_dim1 = 64  (bits 111:80 lo)
    g1[3] = (int)(64u << 16);                            // tile_dim0   = 64  (bits 127:112)
    g1[4] = 64;                                          // tile_dim1   = 64  (bits 143:128)
    g1[5] = 64;                                          // tensor_dim0_stride = 64 (bits 207:160)
    v4i z4 = {0, 0, 0, 0};                               // groups 2/3 unused (2-D)
    v8i z8 = {0, 0, 0, 0, 0, 0, 0, 0};                   // extra group (clang-23 form)
    __builtin_amdgcn_tensor_load_to_lds(g0, g1, z4, z4, z8, 0);
}

__device__ __forceinline__ void tdm_wait0() {
#if __has_builtin(__builtin_amdgcn_s_wait_tensorcnt)
    __builtin_amdgcn_s_wait_tensorcnt(0);
#else
    asm volatile("s_wait_tensorcnt 0" ::: "memory");
#endif
}
#endif

// ---- unified tile-copy frontend: TDM (wave0 programs DMA) or per-lane async
__device__ __forceinline__ void issue_tile_copy(const float* gk, const float* gv,
                                                float* rk, float* rv,
                                                int tid, int wave) {
#if USE_TDM
    if (wave == 0) {
        tdm_load_tile(gk, rk);
        tdm_load_tile(gv, rv);
    }
#else
    const f32x4* gk4 = (const f32x4*)gk;
    const f32x4* gv4 = (const f32x4*)gv;
    f32x4* rk4 = (f32x4*)rk;
    f32x4* rv4 = (f32x4*)rv;
    #pragma unroll
    for (int i = 0; i < 4; ++i) {
        async_copy_b128(&gk4[tid + i * 256], &rk4[tid + i * 256]);
        async_copy_b128(&gv4[tid + i * 256], &rv4[tid + i * 256]);
    }
#endif
}

__device__ __forceinline__ void wait_tile_copy(int wave) {
#if USE_TDM
    if (wave == 0) tdm_wait0();
#else
    async_wait0();
#endif
}

// ---------------- fast exp2 (raw v_exp_f32; args are always <= 0) ----------
__device__ __forceinline__ float fast_exp2(float x) {
#if __has_builtin(__builtin_amdgcn_exp2f)
    return __builtin_amdgcn_exp2f(x);
#else
    return exp2f(x);
#endif
}

// ------------- 16-lane xor-butterfly max via v_permlane16_b32 --------------
__device__ __forceinline__ float lanehalf_max(float v) {
#if __has_builtin(__builtin_amdgcn_permlane16)
    unsigned u, p;
    u = __float_as_uint(v);
    p = __builtin_amdgcn_permlane16(u, u, 0x67452301u, 0xEFCDAB89u, false, false);
    v = fmaxf(v, __uint_as_float(p)); u = __float_as_uint(v);
    p = __builtin_amdgcn_permlane16(u, u, 0x54761032u, 0xDCFE98BAu, false, false);
    v = fmaxf(v, __uint_as_float(p)); u = __float_as_uint(v);
    p = __builtin_amdgcn_permlane16(u, u, 0x32107654u, 0xBA98FEDCu, false, false);
    v = fmaxf(v, __uint_as_float(p)); u = __float_as_uint(v);
    p = __builtin_amdgcn_permlane16(u, u, 0xFEDCBA98u, 0x76543210u, false, false);
    v = fmaxf(v, __uint_as_float(p));
#else
    #pragma unroll
    for (int off = 1; off < 16; off <<= 1)
        v = fmaxf(v, __shfl_xor(v, off, 32));
#endif
    return v;
}

__launch_bounds__(256, 1)
__global__ void fa_fwd_kernel(const float* __restrict__ Q,
                              const float* __restrict__ K,
                              const float* __restrict__ V,
                              float* __restrict__ O)
{
    // Double-buffered raw fp32 tiles (DMA landing zone) + f16 staging.
    __shared__ float    kraw[2][KT * D_DIM] __attribute__((aligned(16)));
    __shared__ float    vraw[2][KT * D_DIM] __attribute__((aligned(16)));
    __shared__ _Float16 ksh[KT][D_DIM]      __attribute__((aligned(32)));
    __shared__ _Float16 vshT[D_DIM][KT]     __attribute__((aligned(32)));
    __shared__ _Float16 pbuf[NWAVE][16][KT] __attribute__((aligned(32)));

    const int tid  = threadIdx.x;
    const int wave = tid >> 5;
    const int lane = tid & 31;
    const int hb   = lane >> 4;    // 0: lanes 0-15, 1: lanes 16-31
    const int ln   = lane & 15;

    const int bh    = blockIdx.y;
    const int qrow0 = blockIdx.x * QT_BLK + wave * 16;

    const float* qbase = Q + (size_t)bh * S_LEN * D_DIM;
    const float* kbase = K + (size_t)bh * S_LEN * D_DIM;
    const float* vbase = V + (size_t)bh * S_LEN * D_DIM;
    float*       obase = O + (size_t)bh * S_LEN * D_DIM;

    // ---- Q tile as two A-fragments (16x32 f16); fold 1/8 * log2(e).
    const float qscale = 0.125f * 1.44269504088896341f;
    const int   mrowA  = ln;
    const int   kbA    = hb * 8;
    v16h aq[2];
    #pragma unroll
    for (int c = 0; c < 2; ++c) {
        const float* qr = qbase + (size_t)(qrow0 + mrowA) * D_DIM + c * 32 + kbA;
        #pragma unroll
        for (int e = 0; e < 8; ++e) aq[c][e]     = (_Float16)(qr[e]      * qscale);
        #pragma unroll
        for (int e = 0; e < 8; ++e) aq[c][8 + e] = (_Float16)(qr[16 + e] * qscale);
    }

    v16h bone;                         // all-ones B fragment (row-sum WMMA)
    #pragma unroll
    for (int e = 0; e < 16; ++e) bone[e] = (_Float16)1.0f;

    v8f o_acc[4];
    #pragma unroll
    for (int t = 0; t < 4; ++t) o_acc[t] = {};
    float mrow[8], lrow[8];
    #pragma unroll
    for (int r = 0; r < 8; ++r) { mrow[r] = -1e30f; lrow[r] = 0.0f; }

    // ---- prologue: copy tile 0 and make it visible to the whole block
    issue_tile_copy(kbase, vbase, kraw[0], vraw[0], tid, wave);
    wait_tile_copy(wave);
    __syncthreads();

    for (int kt = 0; kt < N_TILES; ++kt) {
        const int cur = kt & 1;

        // ---- convert raw slice (fp32 -> f16 staging; K row-major, V^T)
        {
            const f32x4* kr = (const f32x4*)kraw[cur];
            const f32x4* vr = (const f32x4*)vraw[cur];
            #pragma unroll
            for (int i = 0; i < 4; ++i) {
                f32x4 kv = kr[tid + i * 256];
                f32x4 vv = vr[tid + i * 256];
                int base = 4 * (tid + i * 256);
                int key  = base >> 6;
                int d    = base & 63;
                #pragma unroll
                for (int j = 0; j < 4; ++j) {
                    ksh[key][d + j]  = (_Float16)kv[j];
                    vshT[d + j][key] = (_Float16)vv[j];
                }
            }
        }
        // ---- kick off copy of next tile into the other buffer
        if (kt + 1 < N_TILES) {
            issue_tile_copy(kbase + (size_t)(kt + 1) * KT * D_DIM,
                            vbase + (size_t)(kt + 1) * KT * D_DIM,
                            kraw[cur ^ 1], vraw[cur ^ 1], tid, wave);
        }
        __syncthreads();   // f16 tiles ready for all waves

        // ---- scores S[16q x 64k]: 4 groups x 2 accumulating WMMAs
        v8f sc[4];
        #pragma unroll
        for (int g = 0; g < 4; ++g) {
            v8f c = {};
            #pragma unroll
            for (int cc = 0; cc < 2; ++cc) {
                v8h blo = *(const v8h*)&ksh[g * 16 + ln][cc * 32 + hb * 8];
                v8h bhi = *(const v8h*)&ksh[g * 16 + ln][cc * 32 + 16 + hb * 8];
                v16h b = __builtin_shufflevector(blo, bhi, 0,1,2,3,4,5,6,7,8,9,10,11,12,13,14,15);
                c = WMMA_F16(aq[cc], b, c);
            }
            sc[g] = c;
        }

        // ---- online softmax: permlane max butterfly; exp; P -> LDS
        float alpha[8];
        #pragma unroll
        for (int r = 0; r < 8; ++r) {
            float vmax = fmaxf(fmaxf(sc[0][r], sc[1][r]), fmaxf(sc[2][r], sc[3][r]));
            vmax = lanehalf_max(vmax);
            float mnew = fmaxf(mrow[r], vmax);
            alpha[r] = fast_exp2(mrow[r] - mnew);
            mrow[r]  = mnew;
            const int rowM = r + hb * 8;
            #pragma unroll
            for (int g = 0; g < 4; ++g) {
                float p = fast_exp2(sc[g][r] - mnew);
                pbuf[wave][rowM][g * 16 + ln] = (_Float16)p;
            }
            #pragma unroll
            for (int t = 0; t < 4; ++t) o_acc[t][r] *= alpha[r];
        }
        __syncthreads();

        // ---- rebuild P as two A-fragments (keys 0-31, 32-63)
        const _Float16* pr = &pbuf[wave][mrowA][0];
        v8h lo0 = *(const v8h*)&pr[kbA];
        v8h hi0 = *(const v8h*)&pr[kbA + 16];
        v8h lo1 = *(const v8h*)&pr[32 + kbA];
        v8h hi1 = *(const v8h*)&pr[32 + kbA + 16];
        v16h ap0 = __builtin_shufflevector(lo0, hi0, 0,1,2,3,4,5,6,7,8,9,10,11,12,13,14,15);
        v16h ap1 = __builtin_shufflevector(lo1, hi1, 0,1,2,3,4,5,6,7,8,9,10,11,12,13,14,15);

        // ---- row sums on the matrix pipe: sums = P @ ones
        v8f sums = {};
        sums = WMMA_F16(ap0, bone, sums);
        sums = WMMA_F16(ap1, bone, sums);
        #pragma unroll
        for (int r = 0; r < 8; ++r) lrow[r] = lrow[r] * alpha[r] + sums[r];

        // ---- O += P(16x64) @ V(64x64)
        #pragma unroll
        for (int t = 0; t < 4; ++t) {
            v8h vl0 = *(const v8h*)&vshT[t * 16 + ln][hb * 8];
            v8h vh0 = *(const v8h*)&vshT[t * 16 + ln][16 + hb * 8];
            v8h vl1 = *(const v8h*)&vshT[t * 16 + ln][32 + hb * 8];
            v8h vh1 = *(const v8h*)&vshT[t * 16 + ln][48 + hb * 8];
            v16h bv0 = __builtin_shufflevector(vl0, vh0, 0,1,2,3,4,5,6,7,8,9,10,11,12,13,14,15);
            v16h bv1 = __builtin_shufflevector(vl1, vh1, 0,1,2,3,4,5,6,7,8,9,10,11,12,13,14,15);
            o_acc[t] = WMMA_F16(ap0, bv0, o_acc[t]);
            o_acc[t] = WMMA_F16(ap1, bv1, o_acc[t]);
        }
        __syncthreads();       // everyone done with ksh/vshT/raw before reuse

        if (kt + 1 < N_TILES) {
            wait_tile_copy(wave);   // next raw tile landed (wave0 for TDM)
            __syncthreads();        // publish to all waves
        }
    }

    // ---- normalize and write out (C-layout -> [row][dim])
    #pragma unroll
    for (int r = 0; r < 8; ++r) {
        float inv = 1.0f / lrow[r];
        float* orow = obase + (size_t)(qrow0 + r + hb * 8) * D_DIM;
        #pragma unroll
        for (int t = 0; t < 4; ++t)
            orow[t * 16 + ln] = o_acc[t][r] * inv;
    }
}

extern "C" void kernel_launch(void* const* d_in, const int* in_sizes, int n_in,
                              void* d_out, int out_size, void* d_ws, size_t ws_size,
                              hipStream_t stream) {
    const float* Q = (const float*)d_in[0];
    const float* K = (const float*)d_in[1];
    const float* V = (const float*)d_in[2];
    float*       O = (float*)d_out;

    const int bh_count = in_sizes[0] / (S_LEN * D_DIM);   // batch*heads = 32
    dim3 grid(S_LEN / QT_BLK, bh_count);
    dim3 block(256);
    fa_fwd_kernel<<<grid, block, 0, stream>>>(Q, K, V, O);
}
